// SelfAttention_66271345377910
// MI455X (gfx1250) — matrix operand order repaired and checked by reading, not compiled, not run on hardware
//
#include <hip/hip_runtime.h>

// ---------------------------------------------------------------------------
// Causal linear attention (bucketed, blindspot) for MI455X / gfx1250.
// fp32 I/O; bulk GEMMs run as bf16 WMMA (v_wmma_f32_16x16x32_bf16) with fp32
// accumulation; tiles staged to LDS by the Tensor Data Mover (TDM) with
// double buffering + s_wait_tensorcnt; softmax/elu/cumsum stay in fp32.
// ---------------------------------------------------------------------------

typedef __attribute__((ext_vector_type(16))) __bf16    v16bf;
typedef __attribute__((ext_vector_type(8)))  __bf16    v8bf;
typedef __attribute__((ext_vector_type(8)))  float     v8f;
typedef __attribute__((ext_vector_type(4)))  unsigned  v4u;
typedef __attribute__((ext_vector_type(8)))  unsigned  v8u;

#define WMMA_BF16(a, b, c) \
    __builtin_amdgcn_wmma_f32_16x16x32_bf16(false, (a), false, (b), (short)0, (c), false, false)

__device__ __forceinline__ __bf16 f2bf(float f) {
    union { float f; unsigned u; } x; x.f = f;
    unsigned r = x.u + 0x7FFFu + ((x.u >> 16) & 1u);   // round-to-nearest-even
    union { unsigned short s; __bf16 b; } y;
    y.s = (unsigned short)(r >> 16);
    return y.b;
}

// ---------------------------------------------------------------------------
// TDM: issue a 2D bf16 tile load (tile_d0 elems/row, tile_d1 rows) from
// global memory into LDS, padding each 64B row by 16B -> 80B LDS pitch.
// D# layout per CDNA5 ISA ch.8 (group0 128b, group1 256b; groups 2/3 unused
// for 2D tensors). Issued by one wave; completion via TENSORcnt.
// ---------------------------------------------------------------------------
__device__ __forceinline__ void tdm_load_2d_bf16(
    unsigned lds_byte_off, const void* gsrc,
    unsigned tensor_d0, unsigned tensor_d1,
    unsigned tile_d0, unsigned tile_d1, unsigned stride0_elems)
{
    unsigned long long ga = (unsigned long long)(size_t)gsrc;
    v4u g0;
    g0[0] = 1u;                                          // count=1 (valid), user mode
    g0[1] = lds_byte_off;                                // lds_addr
    g0[2] = (unsigned)ga;                                // global_addr[31:0]
    g0[3] = ((unsigned)(ga >> 32) & 0x01FFFFFFu)         // global_addr[56:32]
          | (2u << 30);                                  // type = 2 ("image")
    v8u g1;
    g1[0] = (1u << 16)                                   // data_size = 2 bytes
          | (1u << 20)                                   // pad_enable
          | (3u << 22)                                   // pad_interval: 16 DWORDs (one row)
          | (3u << 25);                                  // pad_amount: 4 DWORDs (16B)
    g1[1] = (tensor_d0 & 0xFFFFu) << 16;                 // tensor_dim0[15:0]
    g1[2] = (tensor_d0 >> 16) | ((tensor_d1 & 0xFFFFu) << 16);
    g1[3] = (tensor_d1 >> 16) | (tile_d0 << 16);         // tile_dim0
    g1[4] = tile_d1 & 0xFFFFu;                           // tile_dim1 (tile_dim2 = 0)
    g1[5] = stride0_elems;                               // tensor_dim0_stride[31:0]
    g1[6] = 0u;
    g1[7] = 0u;
    asm volatile("tensor_load_to_lds %0, %1" :: "s"(g0), "s"(g1) : "memory");
}

// ===========================================================================
// fp32 -> bf16 bulk convert (one-time; removes conversion from GEMM hot loop)
// ===========================================================================
__global__ __launch_bounds__(256) void cvt_bf16_kernel(
    const float* __restrict__ in, __bf16* __restrict__ out)
{
    const size_t i = ((size_t)blockIdx.x * 256 + threadIdx.x) * 4;
    float4 t = *(const float4*)(in + i);
    out[i + 0] = f2bf(t.x);
    out[i + 1] = f2bf(t.y);
    out[i + 2] = f2bf(t.z);
    out[i + 3] = f2bf(t.w);
}

// ===========================================================================
// Big GEMM: C[M,N] = A[M,K] @ W[N,K]^T, both bf16, K-major ("TN" GEMM).
//   MODE: 0 = scatter fp32 result to (b, h, t, e) head layout
//         1 = rowmajor fp32 result + bias
// Block 128x128, K-step 32, 8 waves (4x2), wave = 32x64 = 2x4 frags.
// A/B tiles DMA'd into double-buffered LDS by the TDM (wave 0 issues),
// overlapped with WMMA on the other buffer.
// ===========================================================================
#define TM 128
#define TN 128
#define LDSS 40   // LDS row pitch in halves (80B = 64B row + 16B TDM pad)

template <int MODE>
__global__ __launch_bounds__(256) void gemm_kernel(
    const __bf16* __restrict__ A, const __bf16* __restrict__ W,
    float* __restrict__ outF, const float* __restrict__ bias,
    int M, int N, int K)
{
    __shared__ __bf16 sA[2][TM * LDSS];
    __shared__ __bf16 sB[2][TN * LDSS];

    const int tid   = threadIdx.x;
    const int wave  = tid >> 5;
    const int lane  = tid & 31;
    const int waveM = wave & 3;      // 4 waves along M
    const int waveN = wave >> 2;     // 2 waves along N
    const int hl    = lane >> 4;
    const int l16   = lane & 15;

    const __bf16* aT = A + (size_t)blockIdx.x * TM * K;   // tile (row0, col0)
    const __bf16* bT = W + (size_t)blockIdx.y * TN * K;

    if (tid < 32) {   // wave 0 primes buffer 0
        tdm_load_2d_bf16((unsigned)(size_t)&sA[0][0], aT, K, M, 32, TM, K);
        tdm_load_2d_bf16((unsigned)(size_t)&sB[0][0], bT, K, N, 32, TN, K);
    }

    v8f acc[2][4] = {};
    int cur = 0;
    for (int k0 = 0; k0 < K; k0 += 32) {
        __builtin_amdgcn_s_wait_tensorcnt(0);   // no-op for waves with cnt==0
        __syncthreads();                        // DMA'd tile visible to all waves
        if (tid < 32 && k0 + 32 < K) {          // prefetch next K-tile into other buffer
            tdm_load_2d_bf16((unsigned)(size_t)&sA[cur ^ 1][0], aT + (k0 + 32), K, M, 32, TM, K);
            tdm_load_2d_bf16((unsigned)(size_t)&sB[cur ^ 1][0], bT + (k0 + 32), K, N, 32, TN, K);
        }

        const __bf16* la = sA[cur];
        const __bf16* lb = sB[cur];
        const int aLo = hl ? 16 : 0;   // A frag: lanes0-15 K0-15, lanes16-31 K16-31
        const int bKo = hl ? 16 : 0;   // B frag: lanes0-15 K0-15, lanes16-31 K16-31

        v16bf afr[2];
        #pragma unroll
        for (int i = 0; i < 2; i++) {
            const __bf16* p = &la[(waveM * 32 + i * 16 + l16) * LDSS];
            ((v8bf*)&afr[i])[0] = *(const v8bf*)(p + aLo);
            ((v8bf*)&afr[i])[1] = *(const v8bf*)(p + aLo + 8);
        }
        #pragma unroll
        for (int j = 0; j < 4; j++) {
            v16bf bfr;
            const __bf16* p = &lb[(waveN * 64 + j * 16 + l16) * LDSS + bKo];
            ((v8bf*)&bfr)[0] = *(const v8bf*)(p);
            ((v8bf*)&bfr)[1] = *(const v8bf*)(p + 8);
            #pragma unroll
            for (int i = 0; i < 2; i++)
                acc[i][j] = WMMA_BF16(afr[i], bfr, acc[i][j]);
        }
        __syncthreads();   // all reads of buf `cur` done before it is re-filled
        cur ^= 1;
    }

    // ---- epilogue: C layout = VGPR r -> row r + 8*hl, lane%16 -> col
    const int m0 = blockIdx.x * TM + waveM * 32;
    const int n0 = blockIdx.y * TN + waveN * 64;
    #pragma unroll
    for (int i = 0; i < 2; i++) {
        #pragma unroll
        for (int j = 0; j < 4; j++) {
            #pragma unroll
            for (int r = 0; r < 8; r++) {
                const int m = m0 + i * 16 + r + hl * 8;
                const int n = n0 + j * 16 + l16;
                const float v = acc[i][j][r];
                if (MODE == 0) {
                    // scatter into (b, h, t, e): m = b*8192+t, n = h*64+e
                    const int b = m >> 13, t = m & 8191, h = n >> 6, e = n & 63;
                    outF[(((size_t)(b * 16 + h) * 8192 + t) << 6) + e] = v;
                } else {
                    outF[(size_t)m * N + n] = v + bias[n];
                }
            }
        }
    }
}

// ===========================================================================
// Feature maps per 64x64 bucket: q softmax rows, psi(k)=elu(k*s)+1 (-> k^T
// bf16 + per-bucket column sums via LDS ds_add_f32), v -> v^T bf16.
// ===========================================================================
__global__ __launch_bounds__(256) void featmap_kernel(
    const float* __restrict__ qf, const float* __restrict__ kf,
    const float* __restrict__ vf,
    __bf16* __restrict__ q_bf, __bf16* __restrict__ kT_bf,
    __bf16* __restrict__ vT_bf, float* __restrict__ ksum)
{
    __shared__ float sK[64];
    const int bucket = blockIdx.x;
    const size_t base = (size_t)bucket * 4096;
    const int tid = threadIdx.x;
    const float SC = 0.35355339059327373f;   // 64^-0.25

    if (tid < 64) sK[tid] = 0.f;
    __syncthreads();

    if (tid < 64) {
        float mx = -1e30f;
        for (int j = 0; j < 64; j++) mx = fmaxf(mx, qf[base + tid * 64 + j] * SC);
        float s = 0.f;
        for (int j = 0; j < 64; j++) s += expf(qf[base + tid * 64 + j] * SC - mx);
        const float inv = 1.f / s;
        for (int j = 0; j < 64; j++)
            q_bf[base + tid * 64 + j] = f2bf(expf(qf[base + tid * 64 + j] * SC - mx) * inv);
    } else if (tid < 128) {
        const int row = tid - 64;
        for (int e = 0; e < 64; e++) {
            const float x = kf[base + row * 64 + e] * SC;
            const float f = (x > 0.f) ? (x + 1.f) : expf(x);   // elu(x)+1
            kT_bf[base + e * 64 + row] = f2bf(f);
            atomicAdd(&sK[e], f);
        }
    } else {
        const int idx0 = tid - 128;
        for (int it = 0; it < 32; it++) {
            const int i = idx0 + it * 128;
            const int n = i >> 6, e = i & 63;
            vT_bf[base + e * 64 + n] = f2bf(vf[base + i]);
        }
    }
    __syncthreads();
    if (tid < 64) ksum[(size_t)bucket * 64 + tid] = sK[tid];
}

// ===========================================================================
// Per-bucket context: ctx[d,e] = sum_j k^T[d,j] * v[j,e]  (64x64x64, WMMA)
// ===========================================================================
__global__ __launch_bounds__(128) void ctx_kernel(
    const __bf16* __restrict__ kT, const __bf16* __restrict__ vT,
    float* __restrict__ ctx_raw)
{
    const int bucket = blockIdx.x;
    const __bf16* A = kT + (size_t)bucket * 4096;   // (d x j) rowmajor
    const __bf16* B = vT + (size_t)bucket * 4096;   // (e x j) rowmajor (K-major for B-frag)
    const int wave = threadIdx.x >> 5, lane = threadIdx.x & 31;
    const int hl = lane >> 4, l16 = lane & 15;

    v8f acc[4] = {};
    #pragma unroll
    for (int ks = 0; ks < 64; ks += 32) {
        v16bf af;
        const __bf16* pa = A + (wave * 16 + l16) * 64 + ks + (hl ? 16 : 0);
        ((v8bf*)&af)[0] = *(const v8bf*)(pa);
        ((v8bf*)&af)[1] = *(const v8bf*)(pa + 8);
        #pragma unroll
        for (int j = 0; j < 4; j++) {
            v16bf bf_;
            const __bf16* pb = B + (j * 16 + l16) * 64 + ks + (hl ? 16 : 0);
            ((v8bf*)&bf_)[0] = *(const v8bf*)(pb);
            ((v8bf*)&bf_)[1] = *(const v8bf*)(pb + 8);
            acc[j] = WMMA_BF16(af, bf_, acc[j]);
        }
    }
    float* out = ctx_raw + (size_t)bucket * 4096;
    #pragma unroll
    for (int j = 0; j < 4; j++)
        #pragma unroll
        for (int r = 0; r < 8; r++)
            out[(wave * 16 + r + hl * 8) * 64 + j * 16 + l16] = acc[j][r];
}

// ===========================================================================
// Bucket-serial cumsum + normalize by (kcum[d]+eps) + one-bucket causal shift.
// Writes ctx^T (e-major) bf16 for the attn GEMM's B-fragments.
// ===========================================================================
__global__ __launch_bounds__(256) void cumsum_kernel(
    const float* __restrict__ ctx_raw, const float* __restrict__ ksum,
    __bf16* __restrict__ ctxT_bf)
{
    const int bh = blockIdx.x;     // 0..31
    __shared__ float kc[64];
    const int tid = threadIdx.x;
    if (tid < 64) kc[tid] = 0.f;
    float pref[16] = {};
    __syncthreads();

    for (int u = 0; u < 128; u++) {
        const size_t rb = ((size_t)bh * 128 + u) * 4096;
        #pragma unroll
        for (int j = 0; j < 16; j++) {
            const int idx = tid + 256 * j;
            const int d = idx >> 6, e = idx & 63;
            const float outv = (u == 0) ? 0.f : pref[j] / (kc[d] + 1e-6f);
            ctxT_bf[rb + e * 64 + d] = f2bf(outv);          // shifted, normalized
            pref[j] += ctx_raw[rb + idx];
        }
        __syncthreads();
        if (tid < 64) kc[tid] += ksum[((size_t)bh * 128 + u) * 64 + tid];
        __syncthreads();
    }
}

// ===========================================================================
// Per-bucket attn: attn[n,e] = sum_d q[n,d] * ctx[d,e]; bf16 out in (b,t,dim).
// ===========================================================================
__global__ __launch_bounds__(128) void attn_kernel(
    const __bf16* __restrict__ q_bf, const __bf16* __restrict__ ctxT,
    __bf16* __restrict__ attn_bf)
{
    const int bucket = blockIdx.x;
    const __bf16* A = q_bf + (size_t)bucket * 4096;   // (n x d) rowmajor
    const __bf16* B = ctxT + (size_t)bucket * 4096;   // (e x d) rowmajor
    const int wave = threadIdx.x >> 5, lane = threadIdx.x & 31;
    const int hl = lane >> 4, l16 = lane & 15;

    v8f acc[4] = {};
    #pragma unroll
    for (int ks = 0; ks < 64; ks += 32) {
        v16bf af;
        const __bf16* pa = A + (wave * 16 + l16) * 64 + ks + (hl ? 16 : 0);
        ((v8bf*)&af)[0] = *(const v8bf*)(pa);
        ((v8bf*)&af)[1] = *(const v8bf*)(pa + 8);
        #pragma unroll
        for (int j = 0; j < 4; j++) {
            v16bf bf_;
            const __bf16* pb = B + (j * 16 + l16) * 64 + ks + (hl ? 16 : 0);
            ((v8bf*)&bf_)[0] = *(const v8bf*)(pb);
            ((v8bf*)&bf_)[1] = *(const v8bf*)(pb + 8);
            acc[j] = WMMA_BF16(af, bf_, acc[j]);
        }
    }
    const int bh = bucket >> 7, u = bucket & 127;
    const int b = bh >> 4, h = bh & 15;
    #pragma unroll
    for (int j = 0; j < 4; j++)
        #pragma unroll
        for (int r = 0; r < 8; r++) {
            const int t = u * 64 + wave * 16 + r + hl * 8;
            const int col = h * 64 + j * 16 + l16;
            attn_bf[((size_t)(b * 8192 + t)) * 1024 + col] = f2bf(acc[j][r]);
        }
}

// ===========================================================================
extern "C" void kernel_launch(void* const* d_in, const int* in_sizes, int n_in,
                              void* d_out, int out_size, void* d_ws, size_t ws_size,
                              hipStream_t stream) {
    const float* x  = (const float*)d_in[0];
    const float* Wq = (const float*)d_in[1];
    const float* Wk = (const float*)d_in[2];
    const float* Wv = (const float*)d_in[3];
    const float* Wo = (const float*)d_in[4];
    const float* bo = (const float*)d_in[5];
    float* out = (float*)d_out;
    char* ws = (char*)d_ws;

    const size_t NE = (size_t)2 * 8192 * 1024;   // 16,777,216 elems per activation
    const size_t NW = (size_t)1024 * 1024;       // elems per weight
    // workspace layout (~345 MB, with aliasing):
    float*  qf      = (float*)(ws);               // 64MB fp32; reused as ctx_raw
    float*  kf      = (float*)(ws + NE * 4);      // 64MB fp32; reused as attn_bf
    float*  vf      = (float*)(ws + NE * 8);      // 64MB fp32
    __bf16* q_bf    = (__bf16*)(ws + NE * 12);    // 32MB
    __bf16* kT_bf   = (__bf16*)(ws + NE * 14);    // 32MB
    __bf16* vT_bf   = (__bf16*)(ws + NE * 16);    // 32MB
    float*  ksum    = (float*)(ws + NE * 18);     // 1MB (4096 buckets x 64)
    char*   p2      = ws + NE * 18 + (size_t)4096 * 64 * 4;
    __bf16* ctxT    = (__bf16*)p2;                // 32MB; earlier reused as x_bf
    __bf16* x_bf    = (__bf16*)p2;                //   (x_bf dead before ctxT written)
    __bf16* wq_bf   = (__bf16*)(p2 + NE * 2);
    __bf16* wk_bf   = wq_bf + NW;
    __bf16* wv_bf   = wk_bf + NW;
    __bf16* wo_bf   = wv_bf + NW;                 // 4 x 2MB
    float*  ctx_raw = qf;
    __bf16* attn_bf = (__bf16*)kf;

    const int M = 16384, N = 1024, K = 1024;
    dim3 gg(M / TM, N / TN);

    // one-time bf16 conversion of GEMM operands
    cvt_bf16_kernel<<<(unsigned)(NE / 1024), 256, 0, stream>>>(x, x_bf);
    cvt_bf16_kernel<<<(unsigned)(NW / 1024), 256, 0, stream>>>(Wq, wq_bf);
    cvt_bf16_kernel<<<(unsigned)(NW / 1024), 256, 0, stream>>>(Wk, wk_bf);
    cvt_bf16_kernel<<<(unsigned)(NW / 1024), 256, 0, stream>>>(Wv, wv_bf);
    cvt_bf16_kernel<<<(unsigned)(NW / 1024), 256, 0, stream>>>(Wo, wo_bf);

    gemm_kernel<0><<<gg, 256, 0, stream>>>(x_bf, wq_bf, qf, nullptr, M, N, K);
    gemm_kernel<0><<<gg, 256, 0, stream>>>(x_bf, wk_bf, kf, nullptr, M, N, K);
    gemm_kernel<0><<<gg, 256, 0, stream>>>(x_bf, wv_bf, vf, nullptr, M, N, K);
    featmap_kernel<<<4096, 256, 0, stream>>>(qf, kf, vf, q_bf, kT_bf, vT_bf, ksum);
    ctx_kernel<<<4096, 128, 0, stream>>>(kT_bf, vT_bf, ctx_raw);
    cumsum_kernel<<<32, 256, 0, stream>>>(ctx_raw, ksum, ctxT);
    attn_kernel<<<4096, 128, 0, stream>>>(q_bf, ctxT, attn_bf);
    gemm_kernel<1><<<gg, 256, 0, stream>>>(attn_bf, wo_bf, out, bo, M, N, K);
}
